// ReAttention_70729521431222
// MI455X (gfx1250) — matrix operand order repaired and checked by reading, not compiled, not run on hardware
//
#include <hip/hip_runtime.h>
#include <hip/hip_bf16.h>
#include <stdint.h>

typedef __bf16 bf16;
typedef __attribute__((ext_vector_type(8)))  bf16  v8bf;
typedef __attribute__((ext_vector_type(16))) bf16  v16bf;
typedef __attribute__((ext_vector_type(8)))  float v8f;

#define NB   8        // batch
#define NN   1024     // tokens
#define NC   768      // channels
#define NH   8        // heads
#define HD   96       // head dim
#define SCALE 0.102062072615966f   // 96^-0.5

// ---------------------------------------------------------------------------
// CDNA5 async global->LDS copy helpers (ASYNCcnt path, ISA ch.10 / 15.18.3).
// A generic pointer in the LDS aperture keeps the LDS byte address in its low
// 32 bits (flat-aperture rule), so we can hand per-lane LDS destinations to
// global_load_async_to_lds_* directly.
// ---------------------------------------------------------------------------
__device__ inline uint32_t lds_addr(const void* p) {
    return (uint32_t)(uintptr_t)p;
}
__device__ inline void async_copy_b128(uint32_t lds_dst, const void* gsrc) {
    asm volatile("global_load_async_to_lds_b128 %0, %1, off"
                 :: "v"(lds_dst), "v"(gsrc) : "memory");
}
__device__ inline void async_copy_b32(uint32_t lds_dst, const void* gsrc) {
    asm volatile("global_load_async_to_lds_b32 %0, %1, off"
                 :: "v"(lds_dst), "v"(gsrc) : "memory");
}
__device__ inline void async_wait0() {
    asm volatile("s_wait_asynccnt 0x0" ::: "memory");
}

// ---------------------------------------------------------------------------
// WMMA fragment gathers from LDS (bf16 16x16x32, wave32 layouts)
// A (16x32 row-major, stride ldsStride bf16):
//   lane L: row = L&15 ; K chunks { kb..kb+7 , 16+kb..16+kb+7 }, kb=(L>>4)*8
//   -> two contiguous 16B ds loads.
// B (32x16): lane L: col = L&15 ; K chunks { kb..kb+7 , 16+kb..16+kb+7 }, kb=(L>>4)*8
//   when the LDS tile is stored column-major [n][k].
// C/D (16x16 f32): elem r: M = (L>>4)*8 + r ; N = L&15.
// ---------------------------------------------------------------------------
__device__ inline v16bf frag_a(const bf16* base, int lane, int ldsStride) {
    const bf16* p = base + (lane & 15) * ldsStride + ((lane >> 4) << 3);
    v8bf lo = *(const v8bf*)p;
    v8bf hi = *(const v8bf*)(p + 16);
    return __builtin_shufflevector(lo, hi, 0,1,2,3,4,5,6,7,8,9,10,11,12,13,14,15);
}
__device__ inline v16bf frag_b(const bf16* baseNK, int lane, int ldsStride) {
    const bf16* p = baseNK + (lane & 15) * ldsStride + ((lane >> 4) << 3);
    v8bf lo = *(const v8bf*)p;
    v8bf hi = *(const v8bf*)(p + 16);
    return __builtin_shufflevector(lo, hi, 0,1,2,3,4,5,6,7,8,9,10,11,12,13,14,15);
}
__device__ inline v8f wmma_bf16(v16bf a, v16bf b, v8f c) {
    return __builtin_amdgcn_wmma_f32_16x16x32_bf16(false, a, false, b, (short)0, c, false, false);
}

// ---------------------------------------------------------------------------
// 0) small converters / init
// ---------------------------------------------------------------------------
__global__ void f32_to_bf16_kernel(const float* __restrict__ src, bf16* __restrict__ dst, int n) {
    int i = blockIdx.x * blockDim.x + threadIdx.x;
    if (i < n) dst[i] = (bf16)src[i];
}
__global__ void zero_stats_kernel(float* s) {
    if (threadIdx.x < 16) s[threadIdx.x] = 0.f;
}

// ---------------------------------------------------------------------------
// 1) per-token 3x3 conv (3ch, 16x16, SAME) for q,k,v -> bf16, (B,H,N,hd) layout
//    image loaded via per-lane async global->LDS scatter (padded layout).
// ---------------------------------------------------------------------------
__global__ __launch_bounds__(256) void conv_qkv_kernel(
    const float* __restrict__ x, const float* __restrict__ qw,
    const float* __restrict__ kw, const float* __restrict__ vw,
    bf16* __restrict__ q, bf16* __restrict__ k, bf16* __restrict__ v)
{
    __shared__ float img[3][18][18];   // zero-padded
    __shared__ float wsh[243];         // qw|kw|vw, each 81 = (3,3,3,3)
    const int token = blockIdx.x;              // 0..8191
    const int b = token >> 10, n = token & 1023;
    const float* xt = x + (size_t)token * NC;
    const int t = threadIdx.x;

    for (int i = t; i < 3 * 18 * 18; i += 256) ((float*)img)[i] = 0.f;
    if (t < 243) wsh[t] = (t < 81) ? qw[t] : (t < 162) ? kw[t - 81] : vw[t - 162];
    __syncthreads();
    for (int i = t; i < NC; i += 256) {
        int ch = i >> 8, rem = i & 255;
        uint32_t dst = lds_addr(&img[ch][(rem >> 4) + 1][(rem & 15) + 1]);
        async_copy_b32(dst, xt + i);
    }
    async_wait0();
    __syncthreads();

    const int yy = t >> 4, xx = t & 15;
    float aq[3] = {0.f,0.f,0.f}, ak[3] = {0.f,0.f,0.f}, av[3] = {0.f,0.f,0.f};
    #pragma unroll
    for (int ic = 0; ic < 3; ic++)
      #pragma unroll
      for (int ky = 0; ky < 3; ky++)
        #pragma unroll
        for (int kx = 0; kx < 3; kx++) {
            float p = img[ic][yy + ky][xx + kx];
            #pragma unroll
            for (int oc = 0; oc < 3; oc++) {
                int wi = ((oc * 3 + ic) * 3 + ky) * 3 + kx;
                aq[oc] = fmaf(wsh[wi],       p, aq[oc]);
                ak[oc] = fmaf(wsh[81 + wi],  p, ak[oc]);
                av[oc] = fmaf(wsh[162 + wi], p, av[oc]);
            }
        }
    #pragma unroll
    for (int oc = 0; oc < 3; oc++) {
        int c = oc * 256 + t;          // flatten (3,16,16)
        int h = c / HD, d = c % HD;
        size_t o = (((size_t)b * NH + h) * NN + n) * HD + d;
        q[o] = (bf16)aq[oc]; k[o] = (bf16)ak[oc]; v[o] = (bf16)av[oc];
    }
}

// ---------------------------------------------------------------------------
// 2) scores S = scale * Q K^T   per (b,h): 1024x1024, K=96.  WMMA bf16.
//    grid (16 ntile, 16 mtile, 64 bh), 128 threads (4 waves), 64x64 tile.
//    Q/K tiles staged with async b128 global->LDS.
// ---------------------------------------------------------------------------
__global__ __launch_bounds__(128) void gemm_scores_kernel(
    const bf16* __restrict__ q, const bf16* __restrict__ k, float* __restrict__ S)
{
    alignas(32) __shared__ bf16 Qs[64 * 96];
    alignas(32) __shared__ bf16 Ks[64 * 96];   // K-tensor rows -> column-contig per N
    const int bh = blockIdx.z;
    const bf16* Qp = q + (size_t)bh * NN * HD + (size_t)blockIdx.y * 64 * HD;
    const bf16* Kp = k + (size_t)bh * NN * HD + (size_t)blockIdx.x * 64 * HD;
    const int t = threadIdx.x, wave = t >> 5, lane = t & 31;

    const uint32_t qs = lds_addr(Qs), ks = lds_addr(Ks);
    for (int off = t * 16; off < 64 * 96 * 2; off += 128 * 16) {   // 16B chunks
        async_copy_b128(qs + off, (const char*)Qp + off);
        async_copy_b128(ks + off, (const char*)Kp + off);
    }
    async_wait0();
    __syncthreads();

    v8f acc[4] = {v8f{}, v8f{}, v8f{}, v8f{}};
    #pragma unroll
    for (int kc = 0; kc < 96; kc += 32) {
        v16bf a = frag_a(Qs + (wave * 16) * 96 + kc, lane, 96);
        #pragma unroll
        for (int nt = 0; nt < 4; nt++) {
            v16bf bb = frag_b(Ks + (nt * 16) * 96 + kc, lane, 96);
            acc[nt] = wmma_bf16(a, bb, acc[nt]);
        }
    }
    float* Sp = S + (size_t)bh * NN * NN
                  + (size_t)(blockIdx.y * 64 + wave * 16) * NN + blockIdx.x * 64;
    const int ncol = lane & 15, mbase = (lane >> 4) * 8;
    #pragma unroll
    for (int nt = 0; nt < 4; nt++)
        #pragma unroll
        for (int r = 0; r < 8; r++)
            Sp[(size_t)(mbase + r) * NN + nt * 16 + ncol] = acc[nt][r] * SCALE;
}

// ---------------------------------------------------------------------------
// 3) row softmax over 1024 (one row per 256-thread block, 4 elems/thread)
// ---------------------------------------------------------------------------
__global__ __launch_bounds__(256) void softmax_kernel(float* __restrict__ S)
{
    float* row = S + (size_t)blockIdx.x * NN;
    const int t = threadIdx.x;
    float v0 = row[t], v1 = row[t + 256], v2 = row[t + 512], v3 = row[t + 768];
    __shared__ float red[256];
    red[t] = fmaxf(fmaxf(v0, v1), fmaxf(v2, v3));
    __syncthreads();
    for (int s = 128; s > 0; s >>= 1) { if (t < s) red[t] = fmaxf(red[t], red[t + s]); __syncthreads(); }
    float m = red[0]; __syncthreads();
    v0 = __expf(v0 - m); v1 = __expf(v1 - m); v2 = __expf(v2 - m); v3 = __expf(v3 - m);
    red[t] = v0 + v1 + v2 + v3;
    __syncthreads();
    for (int s = 128; s > 0; s >>= 1) { if (t < s) red[t] += red[t + s]; __syncthreads(); }
    float inv = 1.f / red[0];
    row[t] = v0 * inv; row[t + 256] = v1 * inv; row[t + 512] = v2 * inv; row[t + 768] = v3 * inv;
}

// ---------------------------------------------------------------------------
// 4) re-attention head mix (8x8) + bias, in place; accumulate BN sum/sumsq
// ---------------------------------------------------------------------------
__global__ __launch_bounds__(256) void reattn_kernel(
    float* __restrict__ A, const float* __restrict__ rw, const float* __restrict__ rb,
    float* __restrict__ stats)
{
    const int b  = blockIdx.x >> 6;
    const int n0 = (blockIdx.x & 63) * 16;
    const int t  = threadIdx.x;
    float w[8][8], bias[8];
    #pragma unroll
    for (int o = 0; o < 8; o++) {
        bias[o] = rb[o];
        #pragma unroll
        for (int i = 0; i < 8; i++) w[o][i] = rw[o * 8 + i];
    }
    float sum[8] = {0}, ssq[8] = {0};
    const size_t base = (size_t)b * NH * NN * NN;
    for (int e = t; e < 16 * NN; e += 256) {
        int n = n0 + (e >> 10), m = e & 1023;
        size_t off = base + (size_t)n * NN + m;
        float a[8];
        #pragma unroll
        for (int i = 0; i < 8; i++) a[i] = A[off + (size_t)i * NN * NN];
        #pragma unroll
        for (int o = 0; o < 8; o++) {
            float r = bias[o];
            #pragma unroll
            for (int i = 0; i < 8; i++) r = fmaf(w[o][i], a[i], r);
            A[off + (size_t)o * NN * NN] = r;
            sum[o] += r; ssq[o] = fmaf(r, r, ssq[o]);
        }
    }
    __shared__ float red[256];
    for (int o = 0; o < 8; o++) {
        red[t] = sum[o]; __syncthreads();
        for (int s = 128; s > 0; s >>= 1) { if (t < s) red[t] += red[t + s]; __syncthreads(); }
        if (t == 0) atomicAdd(&stats[o], red[0]);
        __syncthreads();
        red[t] = ssq[o]; __syncthreads();
        for (int s = 128; s > 0; s >>= 1) { if (t < s) red[t] += red[t + s]; __syncthreads(); }
        if (t == 0) atomicAdd(&stats[8 + o], red[0]);
        __syncthreads();
    }
}

__global__ void bn_finalize_kernel(const float* __restrict__ stats,
                                   const float* __restrict__ gamma,
                                   const float* __restrict__ beta,
                                   float* __restrict__ ab)
{
    int o = threadIdx.x;
    if (o < 8) {
        const float cnt = (float)NB * NN * NN;   // mean over (B, N, M)
        float mean = stats[o] / cnt;
        float var  = stats[8 + o] / cnt - mean * mean;
        float s = gamma[o] * rsqrtf(var + 1e-5f);
        ab[o] = s; ab[8 + o] = beta[o] - mean * s;
    }
}

__global__ __launch_bounds__(256) void bn_apply_kernel(float* __restrict__ A,
                                                       const float* __restrict__ ab)
{
    const size_t total  = (size_t)NB * NH * NN * NN;
    const size_t stride = (size_t)gridDim.x * blockDim.x;
    for (size_t i = (size_t)blockIdx.x * blockDim.x + threadIdx.x; i < total; i += stride) {
        int h = (int)((i >> 20) & 7);            // layout (b,h,n,m), NN*NN = 2^20
        A[i] = fmaf(A[i], ab[h], ab[8 + h]);
    }
}

// ---------------------------------------------------------------------------
// 5) out = attn_norm @ V  per (b,h): 1024x96, K=1024.  WMMA bf16.
//    A staged with f32->bf16 conversion; V staged transposed -> VALU path.
// ---------------------------------------------------------------------------
__global__ __launch_bounds__(128) void gemm_av_kernel(
    const float* __restrict__ A, const bf16* __restrict__ v, bf16* __restrict__ outm)
{
    alignas(32) __shared__ bf16 As[64 * 32];     // [m][k]
    alignas(32) __shared__ bf16 Vs[96 * 32];     // [n][k]  transposed stage
    const int bh = blockIdx.y, b = bh >> 3, h = bh & 7;
    const float* Ap = A + (size_t)bh * NN * NN + (size_t)blockIdx.x * 64 * NN;
    const bf16*  Vp = v + (size_t)bh * NN * HD;
    const int t = threadIdx.x, wave = t >> 5, lane = t & 31;

    v8f acc[6] = {v8f{}, v8f{}, v8f{}, v8f{}, v8f{}, v8f{}};
    for (int kc = 0; kc < NN; kc += 32) {
        __syncthreads();
        for (int i = t; i < 64 * 32; i += 128) {            // f32 -> bf16 stage
            int r = i >> 5, c = i & 31;
            As[i] = (bf16)Ap[(size_t)r * NN + kc + c];
        }
        for (int i = t; i < 32 * 96; i += 128) {            // transpose stage
            int kk = i / 96, d = i % 96;
            Vs[d * 32 + kk] = Vp[(size_t)(kc + kk) * HD + d];
        }
        if (kc + 32 < NN) __builtin_prefetch(Ap + (size_t)(kc + 32), 0, 0);
        __syncthreads();
        v16bf a = frag_a(As + (wave * 16) * 32, lane, 32);
        #pragma unroll
        for (int nt = 0; nt < 6; nt++) {
            v16bf bb = frag_b(Vs + (nt * 16) * 32, lane, 32);
            acc[nt] = wmma_bf16(a, bb, acc[nt]);
        }
    }
    const int ncol = lane & 15, mbase = (lane >> 4) * 8;
    #pragma unroll
    for (int nt = 0; nt < 6; nt++)
        #pragma unroll
        for (int r = 0; r < 8; r++) {
            int m = blockIdx.x * 64 + wave * 16 + mbase + r;
            int d = nt * 16 + ncol;
            outm[((size_t)b * NN + m) * NC + h * HD + d] = (bf16)acc[nt][r];
        }
}

// ---------------------------------------------------------------------------
// 6) x_proj = out @ pw^T + pb : (8192x768)@(768x768).  WMMA bf16.
//    A/B tiles staged with async b128 global->LDS each K step.
// ---------------------------------------------------------------------------
__global__ __launch_bounds__(128) void gemm_proj_kernel(
    const bf16* __restrict__ Am, const bf16* __restrict__ Bw,
    const float* __restrict__ pb, float* __restrict__ Xp)
{
    alignas(32) __shared__ bf16 As[64 * 32];
    alignas(32) __shared__ bf16 Bs[64 * 32];   // pw rows (o) -> column-contig per N
    const bf16* Ap = Am + (size_t)blockIdx.y * 64 * NC;
    const bf16* Bp = Bw + (size_t)blockIdx.x * 64 * NC;
    const int t = threadIdx.x, wave = t >> 5, lane = t & 31;
    const uint32_t as = lds_addr(As), bs = lds_addr(Bs);

    v8f acc[4] = {v8f{}, v8f{}, v8f{}, v8f{}};
    for (int kc = 0; kc < NC; kc += 32) {
        __syncthreads();
        for (int i = t; i < 256; i += 128) {    // 64 rows x four 16B chunks
            int r = i >> 2, cb = (i & 3) * 16;
            uint32_t l = (uint32_t)(r * 64 + cb);
            async_copy_b128(as + l, (const char*)(Ap + (size_t)r * NC + kc) + cb);
            async_copy_b128(bs + l, (const char*)(Bp + (size_t)r * NC + kc) + cb);
        }
        if (kc + 32 < NC) __builtin_prefetch(Ap + (size_t)(kc + 32), 0, 0);
        async_wait0();
        __syncthreads();
        v16bf a = frag_a(As + (wave * 16) * 32, lane, 32);
        #pragma unroll
        for (int nt = 0; nt < 4; nt++) {
            v16bf bb = frag_b(Bs + (nt * 16) * 32, lane, 32);
            acc[nt] = wmma_bf16(a, bb, acc[nt]);
        }
    }
    const int ncol = lane & 15, mbase = (lane >> 4) * 8;
    #pragma unroll
    for (int nt = 0; nt < 4; nt++) {
        int o = blockIdx.x * 64 + nt * 16 + ncol;
        float bias = pb[o];
        #pragma unroll
        for (int r = 0; r < 8; r++) {
            int m = blockIdx.y * 64 + wave * 16 + mbase + r;
            Xp[(size_t)m * NC + o] = acc[nt][r] + bias;
        }
    }
}

// ---------------------------------------------------------------------------
// launch
// ---------------------------------------------------------------------------
extern "C" void kernel_launch(void* const* d_in, const int* in_sizes, int n_in,
                              void* d_out, int out_size, void* d_ws, size_t ws_size,
                              hipStream_t stream) {
    const float* x     = (const float*)d_in[0];
    const float* qw    = (const float*)d_in[1];
    const float* kw    = (const float*)d_in[2];
    const float* vw    = (const float*)d_in[3];
    const float* rw    = (const float*)d_in[4];
    const float* rb    = (const float*)d_in[5];
    const float* gamma = (const float*)d_in[6];
    const float* beta  = (const float*)d_in[7];
    const float* pw    = (const float*)d_in[8];
    const float* pb    = (const float*)d_in[9];

    float* x_proj = (float*)d_out;                           // 8*1024*768
    float* attn   = (float*)d_out + (size_t)NB * NN * NC;    // 8*8*1024*1024 (scratch->attn_norm)

    const size_t QKV = (size_t)NB * NH * NN * HD;            // 6,291,456 elems
    bf16* q    = (bf16*)d_ws;
    bf16* k    = q + QKV;
    bf16* v    = k + QKV;
    bf16* outm = v + QKV;                                    // (B,N,C) bf16
    bf16* pwb  = outm + (size_t)NB * NN * NC;
    float* stats = (float*)(pwb + (size_t)NC * NC);          // 16 sums + 16 affine

    f32_to_bf16_kernel<<<(NC * NC + 255) / 256, 256, 0, stream>>>(pw, pwb, NC * NC);
    zero_stats_kernel<<<1, 32, 0, stream>>>(stats);
    conv_qkv_kernel<<<NB * NN, 256, 0, stream>>>(x, qw, kw, vw, q, k, v);
    gemm_scores_kernel<<<dim3(16, 16, NB * NH), 128, 0, stream>>>(q, k, attn);
    softmax_kernel<<<NB * NH * NN, 256, 0, stream>>>(attn);
    reattn_kernel<<<NB * 64, 256, 0, stream>>>(attn, rw, rb, stats);
    bn_finalize_kernel<<<1, 32, 0, stream>>>(stats, gamma, beta, stats + 16);
    bn_apply_kernel<<<8192, 256, 0, stream>>>(attn, stats + 16);
    gemm_av_kernel<<<dim3(16, NB * NH), 128, 0, stream>>>(attn, v, outm);
    gemm_proj_kernel<<<dim3(12, 128), 128, 0, stream>>>(outm, pwb, pb, x_proj);
}
